// ConstraintDecoderModel_65609920413850
// MI455X (gfx1250) — matrix operand order repaired and checked by reading, not compiled, not run on hardware
//
#include <hip/hip_runtime.h>

// ---------------------------------------------------------------------------
// CDNA5 (gfx1250) implementation of the constraint-decoder transformer.
// All dense linears run on v_wmma_f32_16x16x32_f16 (f16 in, f32 accumulate).
// GEMM: one wave computes a 16x64 output strip (4 WMMA tiles), K-loop
// unrolled x2 with ping-pong load buffers: each half-step converts one
// buffer, issues the other buffer's global loads, then fires 4 WMMAs, so
// XDL work hides memory latency without register-rotation copies.
// ---------------------------------------------------------------------------

typedef __attribute__((ext_vector_type(16))) _Float16 v16h;
typedef __attribute__((ext_vector_type(8)))  float    v8f;

#define DM     512      // d_model
#define BB     512      // batch
#define HEADS  8
#define HD     64
#define LCTX   48       // 16 src + 32 tgt
#define LDEC   100      // 4 * 25
#define NC     25

__device__ __forceinline__ void cvt16(v16h& d, const float4& x0, const float4& x1,
                                      const float4& x2, const float4& x3) {
  d[0]=(_Float16)x0.x;  d[1]=(_Float16)x0.y;  d[2]=(_Float16)x0.z;  d[3]=(_Float16)x0.w;
  d[4]=(_Float16)x1.x;  d[5]=(_Float16)x1.y;  d[6]=(_Float16)x1.z;  d[7]=(_Float16)x1.w;
  d[8]=(_Float16)x2.x;  d[9]=(_Float16)x2.y;  d[10]=(_Float16)x2.z; d[11]=(_Float16)x2.w;
  d[12]=(_Float16)x3.x; d[13]=(_Float16)x3.y; d[14]=(_Float16)x3.z; d[15]=(_Float16)x3.w;
}

__device__ __forceinline__ void load_frags(const float* __restrict__ Ap,
                                           const float* __restrict__ Bp,
                                           int K, int k,
                                           float4 ar[4], float4 br[4][4]) {
  ar[0] = *(const float4*)(Ap + k);
  ar[1] = *(const float4*)(Ap + k + 4);
  ar[2] = *(const float4*)(Ap + k + 16);
  ar[3] = *(const float4*)(Ap + k + 20);
#pragma unroll
  for (int t = 0; t < 4; ++t) {
    const float* bp = Bp + (size_t)(16 * t) * K + k;
    br[t][0] = *(const float4*)(bp);
    br[t][1] = *(const float4*)(bp + 4);
    br[t][2] = *(const float4*)(bp + 16);
    br[t][3] = *(const float4*)(bp + 20);
  }
}

// ---------------------------------------------------------------------------
// Y[m,n] = act( bias[n] + sum_k X[m,k] * W[n,k] )
// One wave -> 16 rows x 64 cols (4 accumulators).  M mult of 16, N mult of
// 64, K mult of 64.  Fragment layouts follow CDNA5 ISA 7.12.2.
// act: 0 = none, 1 = relu, 2 = leaky_relu(0.01)
// ---------------------------------------------------------------------------
__global__ void __launch_bounds__(32)
gemm_wmma_kernel(const float* __restrict__ X, const float* __restrict__ W,
                 const float* __restrict__ bias, float* __restrict__ Y,
                 int M, int N, int K, int act) {
  const int lane = threadIdx.x & 31;
  const int n0   = blockIdx.x * 64;
  const int m0   = blockIdx.y * 16;
  const int half = lane >> 4;      // 0: lanes 0-15, 1: lanes 16-31
  const int l16  = lane & 15;

  // A lane(l16, half): row m0+l16, K offsets {half*8 + 0..7, half*8 + 16..23}
  // B lane(l16, half): col n0+16t+l16, K offsets {half*8 + 0..7, half*8 + 16..23}
  const float* Ap = X + (size_t)(m0 + l16) * K + half * 8;
  const float* Bp = W + (size_t)(n0 + l16) * K + half * 8;

  v8f acc[4];
#pragma unroll
  for (int t = 0; t < 4; ++t) {
    const float bn = bias ? bias[n0 + 16 * t + l16] : 0.0f;
#pragma unroll
    for (int r = 0; r < 8; ++r) acc[t][r] = bn;
  }

  float4 arA[4], brA[4][4];   // ping buffer
  float4 arB[4], brB[4][4];   // pong buffer
  load_frags(Ap, Bp, K, 0, arA, brA);

  for (int k = 0; k < K; k += 64) {
    // ---- half-step 0: convert A, load B (k+32), WMMA on A ----
    v16h a0;
    cvt16(a0, arA[0], arA[1], arA[2], arA[3]);
    v16h bf0[4];
#pragma unroll
    for (int t = 0; t < 4; ++t)
      cvt16(bf0[t], brA[t][0], brA[t][1], brA[t][2], brA[t][3]);

    load_frags(Ap, Bp, K, k + 32, arB, brB);     // k+32 < K since K % 64 == 0

#pragma unroll
    for (int t = 0; t < 4; ++t)
      acc[t] = __builtin_amdgcn_wmma_f32_16x16x32_f16(
          false, a0, false, bf0[t], (short)0, acc[t], false, false);

    // ---- half-step 1: convert B, load A (k+64), WMMA on B ----
    v16h a1;
    cvt16(a1, arB[0], arB[1], arB[2], arB[3]);
    v16h bf1[4];
#pragma unroll
    for (int t = 0; t < 4; ++t)
      cvt16(bf1[t], brB[t][0], brB[t][1], brB[t][2], brB[t][3]);

    if (k + 64 < K) {
      __builtin_prefetch(Ap + k + 96, 0, 1);     // global_prefetch_b8
      load_frags(Ap, Bp, K, k + 64, arA, brA);
    }

#pragma unroll
    for (int t = 0; t < 4; ++t)
      acc[t] = __builtin_amdgcn_wmma_f32_16x16x32_f16(
          false, a1, false, bf1[t], (short)0, acc[t], false, false);
  }

#pragma unroll
  for (int t = 0; t < 4; ++t) {
    if (act == 1) {
#pragma unroll
      for (int r = 0; r < 8; ++r) acc[t][r] = fmaxf(acc[t][r], 0.0f);
    } else if (act == 2) {
#pragma unroll
      for (int r = 0; r < 8; ++r)
        acc[t][r] = acc[t][r] > 0.0f ? acc[t][r] : 0.01f * acc[t][r];
    }
  }

  const int mbase = m0 + half * 8;
#pragma unroll
  for (int t = 0; t < 4; ++t)
#pragma unroll
    for (int r = 0; r < 8; ++r)
      Y[(size_t)(mbase + r) * N + n0 + 16 * t + l16] = acc[t][r];
}

// Scalar GEMM for tiny N (head outputs N=4 / N=5). No activation.
__global__ void gemm_small_kernel(const float* __restrict__ X, const float* __restrict__ W,
                                  const float* __restrict__ bias, float* __restrict__ Y,
                                  int M, int N, int K) {
  int idx = blockIdx.x * blockDim.x + threadIdx.x;
  if (idx >= M * N) return;
  int n = idx % N, m = idx / N;
  const float* xp = X + (size_t)m * K;
  const float* wp = W + (size_t)n * K;
  float s = bias[n];
  for (int k = 0; k < K; ++k) s = fmaf(xp[k], wp[k], s);
  Y[idx] = s;
}

// ---------------------------------------------------------------------------
// Online-softmax attention: one thread per (q, b, h).  hd = 64, scale = 1/8.
// Q/K/V layout: (L, B, 512) with head h at columns [h*64, h*64+64).
// ---------------------------------------------------------------------------
__global__ void attention_kernel(const float* __restrict__ Q, const float* __restrict__ Kb,
                                 const float* __restrict__ Vb, float* __restrict__ O,
                                 const unsigned char* __restrict__ pad,
                                 int Lq, int Lk, int B) {
  int idx = blockIdx.x * blockDim.x + threadIdx.x;
  int total = Lq * B * HEADS;
  if (idx >= total) return;
  int h = idx % HEADS;
  int t = idx / HEADS;
  int b = t % B;
  int qi = t / B;

  const float* qp = Q + ((size_t)qi * B + b) * DM + h * HD;
  float qv[HD];
#pragma unroll
  for (int d = 0; d < HD; ++d) qv[d] = qp[d];

  float m = -3.0e38f, l = 0.0f;
  float acc[HD];
#pragma unroll
  for (int d = 0; d < HD; ++d) acc[d] = 0.0f;

  for (int ki = 0; ki < Lk; ++ki) {
    if (pad[b * Lk + ki]) continue;
    const float* kp = Kb + ((size_t)ki * B + b) * DM + h * HD;
    float s = 0.0f;
#pragma unroll
    for (int d = 0; d < HD; ++d) s = fmaf(qv[d], kp[d], s);
    s *= 0.125f;                      // 1/sqrt(64)
    float mn = fmaxf(m, s);
    float corr = __expf(m - mn);
    float p    = __expf(s - mn);
    l = l * corr + p;
    const float* vp = Vb + ((size_t)ki * B + b) * DM + h * HD;
#pragma unroll
    for (int d = 0; d < HD; ++d) acc[d] = acc[d] * corr + p * vp[d];
    m = mn;
  }
  float inv = 1.0f / l;
  float* op = O + ((size_t)qi * B + b) * DM + h * HD;
#pragma unroll
  for (int d = 0; d < HD; ++d) op[d] = acc[d] * inv;
}

// ---------------------------------------------------------------------------
// out = LayerNorm(resid + branch) * g + b       (one wave32 per row of 512)
// ---------------------------------------------------------------------------
__global__ void layernorm_kernel(const float* __restrict__ resid, const float* __restrict__ branch,
                                 const float* __restrict__ g, const float* __restrict__ bb,
                                 float* __restrict__ out, int rows) {
  int row = blockIdx.x * 8 + (threadIdx.x >> 5);
  if (row >= rows) return;
  int lane = threadIdx.x & 31;
  const float* rp = resid  + (size_t)row * DM;
  const float* bp = branch + (size_t)row * DM;
  float x[16], s = 0.0f, s2 = 0.0f;
#pragma unroll
  for (int i = 0; i < 16; ++i) {
    float v = rp[lane + 32 * i] + bp[lane + 32 * i];
    x[i] = v; s += v; s2 += v * v;
  }
#pragma unroll
  for (int off = 16; off > 0; off >>= 1) {
    s  += __shfl_xor(s,  off, 32);
    s2 += __shfl_xor(s2, off, 32);
  }
  float mean = s * (1.0f / DM);
  float var  = s2 * (1.0f / DM) - mean * mean;
  float inv  = rsqrtf(var + 1e-5f);
  float* op = out + (size_t)row * DM;
#pragma unroll
  for (int i = 0; i < 16; ++i) {
    int d = lane + 32 * i;
    op[d] = (x[i] - mean) * inv * g[d] + bb[d];
  }
}

// ctx[l,b,:] = (l<16 ? src_e[l,b,:]+oot[0] : tgt_e[l-16,b,:]+oot[1])
__global__ void build_context_kernel(const float* __restrict__ src_e, const float* __restrict__ tgt_e,
                                     const float* __restrict__ oot, float* __restrict__ ctx) {
  int idx = blockIdx.x * blockDim.x + threadIdx.x;
  if (idx >= LCTX * BB * DM) return;
  int d = idx & 511, b = (idx >> 9) & 511, l = idx >> 18;
  float v = (l < 16) ? src_e[((size_t)l * BB + b) * DM + d] + oot[d]
                     : tgt_e[((size_t)(l - 16) * BB + b) * DM + d] + oot[DM + d];
  ctx[idx] = v;
}

// ctx_pad (B,48) = concat(src_pad, tgt_pad); dec_pad (B,100) = repeat(tgt_c_pad.T, 4)
__global__ void build_pads_kernel(const unsigned char* __restrict__ src_pad,
                                  const unsigned char* __restrict__ tgt_pad,
                                  const unsigned char* __restrict__ tgtc_pad,
                                  unsigned char* __restrict__ ctx_pad,
                                  unsigned char* __restrict__ dec_pad) {
  int idx = blockIdx.x * blockDim.x + threadIdx.x;
  if (idx < BB * LCTX) {
    int b = idx / LCTX, ki = idx % LCTX;
    ctx_pad[idx] = (ki < 16) ? src_pad[b * 16 + ki] : tgt_pad[b * 32 + (ki - 16)];
  } else if (idx < BB * LCTX + BB * LDEC) {
    int j = idx - BB * LCTX;
    int b = j / LDEC, p = j % LDEC, c = p >> 2;
    dec_pad[j] = tgtc_pad[c * BB + b];
  }
}

// Decoder input: embedding gathers + relpos + cidx + sinusoidal PE.
__global__ void build_se_kernel(const float* __restrict__ src_e, const int* __restrict__ tgt_c,
                                const float* __restrict__ type_emb, const float* __restrict__ dir_emb,
                                const float* __restrict__ relpos, const float* __restrict__ cidx,
                                float* __restrict__ se) {
  int idx = blockIdx.x * blockDim.x + threadIdx.x;
  if (idx >= LDEC * BB * DM) return;
  int d = idx & 511, b = (idx >> 9) & 511, pos = idx >> 18;
  int c = pos >> 2, j = pos & 3;
  const int* tc = tgt_c + ((size_t)c * BB + b) * 4;
  float base;
  if (j == 0)      base = type_emb[tc[0] * DM + d];
  else if (j == 1) base = src_e[((size_t)tc[1] * BB + b) * DM + d];
  else if (j == 2) base = src_e[((size_t)tc[2] * BB + b) * DM + d];
  else             base = dir_emb[tc[3] * DM + d];
  // sinusoidal PE: div = exp(-(ln 1e4 / D) * (d & ~1))
  float div = __expf((float)(d & ~1) * (-9.210340371976184f / (float)DM));
  float ang = (float)pos * div;
  float pe  = (d & 1) ? __cosf(ang) : __sinf(ang);
  se[idx] = base + relpos[j * DM + d] + cidx[c * DM + d] + pe;
}

// out[(c*B+b)*512+d] = x[((4c+j)*B+b)*512+d]
__global__ void gather_slice_kernel(const float* __restrict__ x, int j, float* __restrict__ out) {
  int idx = blockIdx.x * blockDim.x + threadIdx.x;
  if (idx >= NC * BB * DM) return;
  int d = idx & 511, b = (idx >> 9) & 511, c = idx >> 18;
  out[idx] = x[(((size_t)(4 * c + j)) * BB + b) * DM + d];
}

// ref_sel[k,b,s] = sum_d src_e[s,b,d] * ptr[k,b,d], masked by src_pad
__global__ void refsel_kernel(const float* __restrict__ src_e, const float* __restrict__ ptr,
                              const unsigned char* __restrict__ src_pad, float* __restrict__ out) {
  int idx = blockIdx.x * blockDim.x + threadIdx.x;
  if (idx >= NC * BB * 16) return;
  int s = idx & 15, b = (idx >> 4) & 511, k = idx >> 13;
  const float* sp = src_e + ((size_t)s * BB + b) * DM;
  const float* pp = ptr   + ((size_t)k * BB + b) * DM;
  float a = 0.0f;
  for (int d = 0; d < DM; ++d) a = fmaf(sp[d], pp[d], a);
  out[idx] = src_pad[b * 16 + s] ? -__builtin_inff() : a;
}

// ---------------------------------------------------------------------------
// Host orchestration
// ---------------------------------------------------------------------------
struct Lin  { const float *b, *w; };
struct Attn { Lin k, o, q, v; };
struct DecL { Attn ca; Lin ff1, ff2;
              const float *ln1b,*ln1g,*ln2b,*ln2g,*ln3b,*ln3g; Attn sa; };
struct EncL { Lin ff1, ff2; const float *ln1b,*ln1g,*ln2b,*ln2g; Attn sa; };

extern "C" void kernel_launch(void* const* d_in, const int* in_sizes, int n_in,
                              void* d_out, int out_size, void* d_ws, size_t ws_size,
                              hipStream_t stream) {
  (void)in_sizes; (void)n_in; (void)out_size; (void)ws_size;
  const float* tgt_e = (const float*)d_in[0];            // (32,512,512)
  const float* src_e = (const float*)d_in[1];            // (16,512,512)
  const float* P     = (const float*)d_in[2];            // params, tree_leaves order
  const int*   tgt_c = (const int*)d_in[3];              // (25,512,4)
  const unsigned char* tgt_pad  = (const unsigned char*)d_in[4];  // (512,32)
  const unsigned char* tgtc_pad = (const unsigned char*)d_in[5];  // (25,512)
  const unsigned char* src_pad  = (const unsigned char*)d_in[6];  // (512,16)
  float* out = (float*)d_out;

  // ---- unpack params (JAX pytree leaf order: dict keys sorted, lists in order)
  size_t off = 0;
  auto take = [&](size_t n) { const float* p = P + off; off += n; return p; };
  auto takeAttn = [&]() {
    Attn a;
    a.k.b = take(512); a.k.w = take(512 * 512);
    a.o.b = take(512); a.o.w = take(512 * 512);
    a.q.b = take(512); a.q.w = take(512 * 512);
    a.v.b = take(512); a.v.w = take(512 * 512);
    return a;
  };
  const float* cidx = take(32 * 512);                    // cidx_emb
  DecL dec[6];
  for (int l = 0; l < 6; ++l) {                          // dec (keys: ca,ff1,ff2,ln1,ln2,ln3,sa)
    dec[l].ca  = takeAttn();
    dec[l].ff1 = { take(2048), take(2048 * 512) };
    dec[l].ff2 = { take(512),  take(512 * 2048) };
    dec[l].ln1b = take(512); dec[l].ln1g = take(512);
    dec[l].ln2b = take(512); dec[l].ln2g = take(512);
    dec[l].ln3b = take(512); dec[l].ln3g = take(512);
    dec[l].sa  = takeAttn();
  }
  const float* dir_emb = take(5 * 512);                  // dir_emb
  Lin dir0 = { take(1024), take(1024 * 512) };           // dir_head[0]
  Lin dir1 = { take(5),    take(5 * 1024) };             // dir_head[1]
  EncL enc[6];
  for (int l = 0; l < 6; ++l) {                          // enc (keys: ff1,ff2,ln1,ln2,sa)
    enc[l].ff1 = { take(2048), take(2048 * 512) };
    enc[l].ff2 = { take(512),  take(512 * 2048) };
    enc[l].ln1b = take(512); enc[l].ln1g = take(512);
    enc[l].ln2b = take(512); enc[l].ln2g = take(512);
    enc[l].sa  = takeAttn();
  }
  const float* oot = take(2 * 512);                      // oot_emb
  Lin ptr0 = { take(1024), take(1024 * 512) };           // pointer_head[0]
  Lin ptr1 = { take(512),  take(512 * 1024) };           // pointer_head[1]
  const float* relpos   = take(4 * 512);                 // relpos_emb
  const float* type_emb = take(4 * 512);                 // type_emb
  Lin typ0 = { take(1024), take(1024 * 512) };           // type_head[0]
  Lin typ1 = { take(4),    take(4 * 1024) };             // type_head[1]

  // ---- workspace arena
  const size_t SLAB = 26214400;                          // 51200*512 floats
  float* mem  = (float*)d_ws;                            // 24576*512
  float* x    = mem  + 12582912;
  float* bq   = x    + SLAB;
  float* bk   = bq   + SLAB;
  float* bv   = bk   + SLAB;
  float* bmid = bv   + SLAB;                             // 12800*2048 chunks
  unsigned char* ctx_pad = (unsigned char*)(bmid + SLAB);
  unsigned char* dec_pad = ctx_pad + BB * LCTX;

  const int Me = LCTX * BB;   // 24576
  const int Md = LDEC * BB;   // 51200
  const int Mh = NC * BB;     // 12800
  const int CHUNK = 12800;

  auto gemm = [&](const float* X_, const Lin& L_, float* Y_, int M, int N, int K, int act) {
    dim3 grid(N / 64, M / 16);
    gemm_wmma_kernel<<<grid, 32, 0, stream>>>(X_, L_.w, L_.b, Y_, M, N, K, act);
  };
  auto attn = [&](const float* Q, const float* K_, const float* V, float* O,
                  const unsigned char* pad, int Lq, int Lk) {
    int tot = Lq * BB * HEADS;
    attention_kernel<<<(tot + 255) / 256, 256, 0, stream>>>(Q, K_, V, O, pad, Lq, Lk, BB);
  };
  auto lnorm = [&](const float* r, const float* br, const float* g, const float* b2,
                   float* o, int rows) {
    layernorm_kernel<<<(rows + 7) / 8, 256, 0, stream>>>(r, br, g, b2, o, rows);
  };
  auto ffn = [&](float* stream_buf, const Lin& f1, const Lin& f2, float* branch, int M) {
    for (int o2 = 0; o2 < M; o2 += CHUNK) {
      int cm = (M - o2 < CHUNK) ? (M - o2) : CHUNK;
      gemm(stream_buf + (size_t)o2 * DM, f1, bmid, cm, 2048, DM, 1);
      gemm(bmid, f2, branch + (size_t)o2 * DM, cm, DM, 2048, 0);
    }
  };

  // ---- inputs
  {
    int t = LCTX * BB * DM;
    build_context_kernel<<<(t + 255) / 256, 256, 0, stream>>>(src_e, tgt_e, oot, mem);
    int tp = BB * (LCTX + LDEC);
    build_pads_kernel<<<(tp + 255) / 256, 256, 0, stream>>>(src_pad, tgt_pad, tgtc_pad,
                                                            ctx_pad, dec_pad);
    int ts = LDEC * BB * DM;
    build_se_kernel<<<(ts + 255) / 256, 256, 0, stream>>>(src_e, tgt_c, type_emb, dir_emb,
                                                          relpos, cidx, x);
  }

  // ---- encoder
  for (int l = 0; l < 6; ++l) {
    EncL& E = enc[l];
    gemm(mem, E.sa.q, bq, Me, DM, DM, 0);
    gemm(mem, E.sa.k, bk, Me, DM, DM, 0);
    gemm(mem, E.sa.v, bv, Me, DM, DM, 0);
    attn(bq, bk, bv, bmid, ctx_pad, LCTX, LCTX);
    gemm(bmid, E.sa.o, bq, Me, DM, DM, 0);
    lnorm(mem, bq, E.ln1g, E.ln1b, mem, Me);
    ffn(mem, E.ff1, E.ff2, bk, Me);
    lnorm(mem, bk, E.ln2g, E.ln2b, mem, Me);
  }

  // ---- decoder
  for (int l = 0; l < 6; ++l) {
    DecL& Dd = dec[l];
    // self-attention
    gemm(x, Dd.sa.q, bq, Md, DM, DM, 0);
    gemm(x, Dd.sa.k, bk, Md, DM, DM, 0);
    gemm(x, Dd.sa.v, bv, Md, DM, DM, 0);
    attn(bq, bk, bv, bmid, dec_pad, LDEC, LDEC);
    gemm(bmid, Dd.sa.o, bq, Md, DM, DM, 0);
    lnorm(x, bq, Dd.ln1g, Dd.ln1b, x, Md);
    // cross-attention
    gemm(x,   Dd.ca.q, bq, Md, DM, DM, 0);
    gemm(mem, Dd.ca.k, bk, Me, DM, DM, 0);
    gemm(mem, Dd.ca.v, bv, Me, DM, DM, 0);
    attn(bq, bk, bv, bmid, ctx_pad, LDEC, LCTX);
    gemm(bmid, Dd.ca.o, bq, Md, DM, DM, 0);
    lnorm(x, bq, Dd.ln2g, Dd.ln2b, x, Md);
    // FFN
    ffn(x, Dd.ff1, Dd.ff2, bk, Md);
    lnorm(x, bk, Dd.ln3g, Dd.ln3b, x, Md);
  }

  // ---- heads
  int tg = Mh * DM;
  // type head: x[0::4] -> leaky -> (25,512,4) at out+0
  gather_slice_kernel<<<(tg + 255) / 256, 256, 0, stream>>>(x, 0, bq);
  gemm(bq, typ0, bmid, Mh, 1024, DM, 2);
  gemm_small_kernel<<<(Mh * 4 + 255) / 256, 256, 0, stream>>>(bmid, typ1.w, typ1.b,
                                                              out, Mh, 4, 1024);
  // pointer head: x[2::4] -> leaky -> (25,512,512) -> einsum -> (25,512,16) at out+51200
  gather_slice_kernel<<<(tg + 255) / 256, 256, 0, stream>>>(x, 2, bq);
  gemm(bq, ptr0, bmid, Mh, 1024, DM, 2);
  gemm(bmid, ptr1, bk, Mh, DM, 1024, 0);
  refsel_kernel<<<(NC * BB * 16 + 255) / 256, 256, 0, stream>>>(src_e, bk, src_pad,
                                                                out + 51200);
  // dir head: x[3::4] -> leaky -> (25,512,5) at out+256000
  gather_slice_kernel<<<(tg + 255) / 256, 256, 0, stream>>>(x, 3, bq);
  gemm(bq, dir0, bmid, Mh, 1024, DM, 2);
  gemm_small_kernel<<<(Mh * 5 + 255) / 256, 256, 0, stream>>>(bmid, dir1.w, dir1.b,
                                                              out + 256000, Mh, 5, 1024);
}
